// MoELayer_56049323213101
// MI455X (gfx1250) — matrix operand order repaired and checked by reading, not compiled, not run on hardware
//
#include <hip/hip_runtime.h>
#include <hip/hip_bf16.h>
#include <math.h>

// Problem constants (match reference)
#define D_  1024
#define H_  2048
#define E_  8
#define K_  2
#define T_  8192                  // B*L tokens
#define MT_ 128                   // row-tile (M) per block
#define CAP_  (T_*K_ + E_*MT_)    // 17408 padded slots (expert segments 128-aligned)
#define CAPT_ (CAP_/MT_)          // 136 row tiles

#define LDA  40                   // LDS stride for A rows  (>=32, mult of 8 -> 16B align)
#define LDBT 40                   // LDS stride for B^T cols (>=32, mult of 8 -> 16B align)

typedef __attribute__((ext_vector_type(16))) __bf16 v16bf;
typedef __attribute__((ext_vector_type(8)))  __bf16 v8bf;
typedef __attribute__((ext_vector_type(8)))  float  v8f;
typedef __attribute__((__vector_size__(16))) int    b128_t;   // async-LDS payload type

#define AS1 __attribute__((address_space(1)))
#define AS3 __attribute__((address_space(3)))

__device__ __forceinline__ __bf16 tobf(float f) { return (__bf16)f; }

#define HAS_ASYNC_LDS (__has_builtin(__builtin_amdgcn_global_load_async_to_lds_b128))

__device__ __forceinline__ void async_copy16(const void* g, void* l) {
#if HAS_ASYNC_LDS
  __builtin_amdgcn_global_load_async_to_lds_b128(
      (AS1 b128_t*)(const AS1 void*)g,
      (AS3 b128_t*)(AS3 void*)l, 0, 0);
#endif
}

__device__ __forceinline__ void wait_asynccnt0() {
#if __has_builtin(__builtin_amdgcn_s_wait_asynccnt)
  __builtin_amdgcn_s_wait_asynccnt(0);
#else
  asm volatile("s_wait_asynccnt 0x0" ::: "memory");
#endif
}

// ---- WMMA fragment builders (ISA 7.12.2 layouts, wave32) -------------------
// A (16x32 bf16): lane half 0 holds K {0..7,16..23}, half 1 holds K {8..15,24..31}
__device__ __forceinline__ v16bf frag_a(const __bf16* sA, int mRow, int half) {
  const __bf16* p = sA + mRow * LDA;
  v8bf lo = *(const v8bf*)(p + half * 8);
  v8bf hi = *(const v8bf*)(p + 16 + half * 8);
  return __builtin_shufflevector(lo, hi, 0, 1, 2, 3, 4, 5, 6, 7,
                                 8, 9, 10, 11, 12, 13, 14, 15);
}
// B (32x16 bf16) from transposed LDS tile sBT[ncol*LDBT + k]:
// column = lane%16; lane half 0 holds K {0..7,16..23}, half 1 holds K {8..15,24..31}
__device__ __forceinline__ v16bf frag_b(const __bf16* sBT, int ncol, int half) {
  const __bf16* p = sBT + ncol * LDBT + half * 8;
  v8bf lo = *(const v8bf*)(p);
  v8bf hi = *(const v8bf*)(p + 16);
  return __builtin_shufflevector(lo, hi, 0, 1, 2, 3, 4, 5, 6, 7,
                                 8, 9, 10, 11, 12, 13, 14, 15);
}

// ---- init: zero histograms, poison slot map --------------------------------
__global__ __launch_bounds__(256) void init_kernel(int* counts, int* token_map) {
  int i = blockIdx.x * 256 + threadIdx.x;
  if (i < E_)  counts[i] = 0;
  if (i < CAP_) token_map[i] = -1;
}

// ---- router: one wave32 per token, top-2 + softmax + histogram -------------
__global__ __launch_bounds__(128)
void router_kernel(const float* __restrict__ x, const float* __restrict__ rw,
                   const float* __restrict__ bias,
                   int* __restrict__ sel_id, float* __restrict__ sel_w,
                   int* __restrict__ counts) {
  int wave = threadIdx.x >> 5, lane = threadIdx.x & 31;
  int t = blockIdx.x * 4 + wave;
  float acc[E_];
#pragma unroll
  for (int e = 0; e < E_; ++e) acc[e] = 0.f;
  for (int kk = 0; kk < D_ / 32; ++kk) {
    int k = kk * 32 + lane;
    float xv = x[(size_t)t * D_ + k];
#pragma unroll
    for (int e = 0; e < E_; ++e) acc[e] += xv * rw[e * D_ + k];
  }
#pragma unroll
  for (int e = 0; e < E_; ++e) {
    float v = acc[e];
#pragma unroll
    for (int off = 16; off; off >>= 1) v += __shfl_xor(v, off, 32);
    acc[e] = v;
  }
  if (lane == 0) {
    float lg[E_];
#pragma unroll
    for (int e = 0; e < E_; ++e) lg[e] = acc[e] + bias[e];
    int i1 = 0;
#pragma unroll
    for (int e = 1; e < E_; ++e) if (lg[e] > lg[i1]) i1 = e;
    int i2 = (i1 == 0) ? 1 : 0;
#pragma unroll
    for (int e = 0; e < E_; ++e) if (e != i1 && lg[e] > lg[i2]) i2 = e;
    float p2 = __expf(lg[i2] - lg[i1]);
    float inv = 1.f / (1.f + p2);
    sel_id[t * 2 + 0] = i1;  sel_w[t * 2 + 0] = inv;
    sel_id[t * 2 + 1] = i2;  sel_w[t * 2 + 1] = p2 * inv;
    atomicAdd(&counts[i1], 1);
    atomicAdd(&counts[i2], 1);
  }
}

// ---- scan: MT_-aligned per-expert segment offsets --------------------------
__global__ void scan_kernel(const int* __restrict__ counts,
                            int* __restrict__ offsets, int* __restrict__ cursors) {
  if (threadIdx.x == 0) {
    int off = 0;
#pragma unroll
    for (int e = 0; e < E_; ++e) { offsets[e] = off; off += (counts[e] + MT_ - 1) & ~(MT_ - 1); }
    offsets[E_] = off;
  }
  if (threadIdx.x < E_) cursors[threadIdx.x] = 0;
}

// ---- scatter tokens into expert segments -----------------------------------
__global__ __launch_bounds__(256)
void scatter_kernel(const int* __restrict__ sel_id, const float* __restrict__ sel_w,
                    const int* __restrict__ offsets, int* __restrict__ cursors,
                    int* __restrict__ token_map, float* __restrict__ weight_map) {
  int t = blockIdx.x * 256 + threadIdx.x;
#pragma unroll
  for (int k = 0; k < K_; ++k) {
    int e = sel_id[t * 2 + k];
    int slot = offsets[e] + atomicAdd(&cursors[e], 1);
    token_map[slot] = t;
    weight_map[slot] = sel_w[t * 2 + k];
  }
}

// ---- up projection + SwiGLU: act = silu(X@W1) * (X@W3), bf16 out -----------
// Block: 256 thr (8 waves). Tile: 128 rows x 64 cols of H. K loop over D.
template <bool MOE>
__global__ __launch_bounds__(256)
void up_act_kernel(const float* __restrict__ x,
                   const float* __restrict__ w1, const float* __restrict__ w3,
                   __bf16* __restrict__ act,
                   const int* __restrict__ offsets, const int* __restrict__ token_map) {
  int rowBase = blockIdx.x * MT_;
  int nBase   = blockIdx.y * 64;
  int e = 0;
  if (MOE) {
    if (rowBase >= offsets[E_]) return;
    while (rowBase >= offsets[e + 1]) ++e;
  }
  const float* W1 = MOE ? w1 + (size_t)e * D_ * H_ : w1;
  const float* W3 = MOE ? w3 + (size_t)e * D_ * H_ : w3;

  __shared__ __bf16 sA  [MT_ * LDA];   // 128 x 32 (row-major, padded)
  __shared__ __bf16 sBT1[64 * LDBT];   // 64 cols x 32 K (transposed)
  __shared__ __bf16 sBT3[64 * LDBT];

  int tid = threadIdx.x;
  // A loader: row = tid/2, 16 consecutive K cols
  int ar = tid >> 1, ac = (tid & 1) * 16;
  int tok = MOE ? token_map[rowBase + ar] : (rowBase + ar);
  // B loader: K-row = tid/8, 8 consecutive N cols (coalesced global read)
  int br = tid >> 3, bc = (tid & 7) * 8;

  int lane = tid & 31, wave = tid >> 5;
  int half = lane >> 4, lm = lane & 15;
  int mT = wave;  // wave w owns rows [w*16, w*16+16)

  v8f accg[4] = {v8f{}, v8f{}, v8f{}, v8f{}};
  v8f accu[4] = {v8f{}, v8f{}, v8f{}, v8f{}};

  for (int k0 = 0; k0 < D_; k0 += 32) {
    // stage A (fp32 -> bf16), 16B vector stores
    {
      v8bf va, vb;
      if (tok >= 0) {
        const float* xp = x + (size_t)tok * D_ + k0 + ac;
#pragma unroll
        for (int j = 0; j < 8; ++j) va[j] = tobf(xp[j]);
#pragma unroll
        for (int j = 0; j < 8; ++j) vb[j] = tobf(xp[8 + j]);
      } else {
#pragma unroll
        for (int j = 0; j < 8; ++j) { va[j] = tobf(0.f); vb[j] = tobf(0.f); }
      }
      *(v8bf*)&sA[ar * LDA + ac]     = va;
      *(v8bf*)&sA[ar * LDA + ac + 8] = vb;
    }
    // stage B1/B3 transposed (coalesced global read, scattered LDS store)
    {
      const float* b1 = W1 + (size_t)(k0 + br) * H_ + nBase + bc;
      const float* b3 = W3 + (size_t)(k0 + br) * H_ + nBase + bc;
#pragma unroll
      for (int j = 0; j < 8; ++j) sBT1[(bc + j) * LDBT + br] = tobf(b1[j]);
#pragma unroll
      for (int j = 0; j < 8; ++j) sBT3[(bc + j) * LDBT + br] = tobf(b3[j]);
      if (k0 + 32 < D_) {
        __builtin_prefetch(b1 + 32 * H_, 0, 0);   // global_prefetch_b8
        __builtin_prefetch(b3 + 32 * H_, 0, 0);
      }
    }
    __syncthreads();

    v16bf a = frag_a(sA, mT * 16 + lm, half);
#pragma unroll
    for (int nt = 0; nt < 4; ++nt) {
      v16bf bg = frag_b(sBT1, nt * 16 + lm, half);
      v16bf bu = frag_b(sBT3, nt * 16 + lm, half);
      accg[nt] = __builtin_amdgcn_wmma_f32_16x16x32_bf16(false, a, false, bg, (short)0, accg[nt], false, false);
      accu[nt] = __builtin_amdgcn_wmma_f32_16x16x32_bf16(false, a, false, bu, (short)0, accu[nt], false, false);
    }
    __syncthreads();
  }

  // epilogue: SiLU(g)*u -> bf16 act[slot, hcol]
#pragma unroll
  for (int nt = 0; nt < 4; ++nt) {
    int col = nBase + nt * 16 + lm;
#pragma unroll
    for (int v = 0; v < 8; ++v) {
      int s = rowBase + mT * 16 + half * 8 + v;   // slot (MOE) or token (shared)
      float g = accg[nt][v], u = accu[nt][v];
      float av = (g / (1.f + __expf(-g))) * u;
      act[(size_t)s * H_ + col] = tobf(av);
    }
  }
}

// ---- down projection: out += cw * (act @ W2) -------------------------------
// Tile 128 rows x 64 cols of D. K loop over H. A (bf16) staged via async-LDS.
template <bool MOE>
__global__ __launch_bounds__(256)
void down_kernel(const __bf16* __restrict__ act, const float* __restrict__ w2,
                 float* __restrict__ out,
                 const int* __restrict__ offsets, const int* __restrict__ token_map,
                 const float* __restrict__ weight_map) {
  int rowBase = blockIdx.x * MT_;
  int nBase   = blockIdx.y * 64;
  int e = 0;
  if (MOE) {
    if (rowBase >= offsets[E_]) return;
    while (rowBase >= offsets[e + 1]) ++e;
  }
  const float* W2 = MOE ? w2 + (size_t)e * H_ * D_ : w2;

  __shared__ __bf16 sA [MT_ * LDA];
  __shared__ __bf16 sBT[64 * LDBT];

  int tid = threadIdx.x;
  int ar = tid >> 1, ac = (tid & 1) * 16;
  int br = tid >> 3, bc = (tid & 7) * 8;

  int lane = tid & 31, wave = tid >> 5;
  int half = lane >> 4, lm = lane & 15;
  int mT = wave;

  v8f acc[4] = {v8f{}, v8f{}, v8f{}, v8f{}};

  const __bf16* arow = act + (size_t)(rowBase + ar) * H_ + ac;

  for (int k0 = 0; k0 < H_; k0 += 32) {
    // stage A (already bf16): async global->LDS copy when available
    {
      const __bf16* ap = arow + k0;
      __bf16* lp = &sA[ar * LDA + ac];
#if HAS_ASYNC_LDS
      async_copy16(ap, lp);
      async_copy16(ap + 8, lp + 8);
#else
      *(v8bf*)lp       = *(const v8bf*)ap;
      *(v8bf*)(lp + 8) = *(const v8bf*)(ap + 8);
#endif
    }
    // stage B transposed (fp32 -> bf16)
    {
      const float* bp = W2 + (size_t)(k0 + br) * D_ + nBase + bc;
#pragma unroll
      for (int j = 0; j < 8; ++j) sBT[(bc + j) * LDBT + br] = tobf(bp[j]);
      if (k0 + 32 < H_) __builtin_prefetch(bp + 32 * D_, 0, 0);
    }
#if HAS_ASYNC_LDS
    wait_asynccnt0();
#endif
    __syncthreads();

    v16bf a = frag_a(sA, mT * 16 + lm, half);
#pragma unroll
    for (int nt = 0; nt < 4; ++nt) {
      v16bf b = frag_b(sBT, nt * 16 + lm, half);
      acc[nt] = __builtin_amdgcn_wmma_f32_16x16x32_bf16(false, a, false, b, (short)0, acc[nt], false, false);
    }
    __syncthreads();
  }

#pragma unroll
  for (int nt = 0; nt < 4; ++nt) {
    int col = nBase + nt * 16 + lm;
#pragma unroll
    for (int v = 0; v < 8; ++v) {
      int s = rowBase + mT * 16 + half * 8 + v;
      if (MOE) {
        int t = token_map[s];
        if (t >= 0) {
          atomicAdd(&out[(size_t)t * D_ + col], acc[nt][v] * weight_map[s]);
        }
      } else {
        out[(size_t)s * D_ + col] = acc[nt][v];
      }
    }
  }
}

// ---- host-side launcher ----------------------------------------------------
extern "C" void kernel_launch(void* const* d_in, const int* in_sizes, int n_in,
                              void* d_out, int out_size, void* d_ws, size_t ws_size,
                              hipStream_t stream) {
  const float* x    = (const float*)d_in[0];
  const float* rw   = (const float*)d_in[1];
  const float* bias = (const float*)d_in[2];
  const float* w1   = (const float*)d_in[3];
  const float* w3   = (const float*)d_in[4];
  const float* w2   = (const float*)d_in[5];
  const float* sw1  = (const float*)d_in[6];
  const float* sw3  = (const float*)d_in[7];
  const float* sw2  = (const float*)d_in[8];
  float* out = (float*)d_out;

  // workspace carve-up
  char* p = (char*)d_ws;
  __bf16* act = (__bf16*)p;              p += (size_t)CAP_ * H_ * sizeof(__bf16);
  int*   sel_id     = (int*)p;           p += (size_t)T_ * K_ * sizeof(int);
  float* sel_w      = (float*)p;         p += (size_t)T_ * K_ * sizeof(float);
  int*   counts     = (int*)p;           p += 64 * sizeof(int);
  int*   cursors    = (int*)p;           p += 64 * sizeof(int);
  int*   offsets    = (int*)p;           p += 64 * sizeof(int);
  int*   token_map  = (int*)p;           p += (size_t)CAP_ * sizeof(int);
  float* weight_map = (float*)p;         p += (size_t)CAP_ * sizeof(float);
  (void)ws_size; (void)in_sizes; (void)n_in; (void)out_size;

  // 1. init routing state
  init_kernel<<<(CAP_ + 255) / 256, 256, 0, stream>>>(counts, token_map);
  // 2. shared expert: up+SwiGLU then down (plain stores initialize d_out)
  up_act_kernel<false><<<dim3(T_ / MT_, H_ / 64), 256, 0, stream>>>(
      x, sw1, sw3, act, nullptr, nullptr);
  down_kernel<false><<<dim3(T_ / MT_, D_ / 64), 256, 0, stream>>>(
      act, sw2, out, nullptr, nullptr, nullptr);
  // 3. router top-2
  router_kernel<<<T_ / 4, 128, 0, stream>>>(x, rw, bias, sel_id, sel_w, counts);
  // 4. segment offsets + scatter
  scan_kernel<<<1, 32, 0, stream>>>(counts, offsets, cursors);
  scatter_kernel<<<T_ / 256, 256, 0, stream>>>(sel_id, sel_w, offsets, cursors,
                                               token_map, weight_map);
  // 5. grouped expert FFN (sparse: only selected expert rows)
  up_act_kernel<true><<<dim3(CAPT_, H_ / 64), 256, 0, stream>>>(
      x, w1, w3, act, offsets, token_map);
  down_kernel<true><<<dim3(CAPT_, D_ / 64), 256, 0, stream>>>(
      act, w2, out, offsets, token_map, weight_map);
}
